// StackedLSTM_8083128451731
// MI455X (gfx1250) — matrix operand order repaired and checked by reading, not compiled, not run on hardware
//
#include <hip/hip_runtime.h>

typedef __attribute__((ext_vector_type(16))) _Float16 v16h;
typedef __attribute__((ext_vector_type(8)))  _Float16 v8h;
typedef __attribute__((ext_vector_type(8)))  float    v8f;

#define LSTM_T   256
#define LSTM_IN  3
#define HP       32      // padded hidden (30 -> 32)
#define GP       128     // padded gates (120 -> 128)
#define KP       64      // padded concat-K per layer
#define ROWS_PER_WG   128
#define LSTM_BATCH    8192

#define NEG_LOG2E   (-1.4426950408889634f)
#define NEG_2LOG2E  (-2.8853900817779268f)

// sigmoid(a + b) with pre-scaled bias bs = b * NEG_LOG2E
__device__ __forceinline__ float sigmoid_b(float a, float bs) {
  float e = __builtin_amdgcn_exp2f(fmaf(a, NEG_LOG2E, bs));
  return __builtin_amdgcn_rcpf(1.0f + e);
}
// tanh(a + b) with pre-scaled bias bs = b * NEG_2LOG2E
__device__ __forceinline__ float tanh_b(float a, float bs) {
  float e = __builtin_amdgcn_exp2f(fmaf(a, NEG_2LOG2E, bs));
  return fmaf(__builtin_amdgcn_rcpf(1.0f + e), 2.0f, -1.0f);
}
__device__ __forceinline__ float fast_tanh(float v) {
  float e = __builtin_amdgcn_exp2f(v * NEG_2LOG2E);
  return fmaf(__builtin_amdgcn_rcpf(1.0f + e), 2.0f, -1.0f);
}
__device__ __forceinline__ float fast_sigmoid(float v) {
  float e = __builtin_amdgcn_exp2f(v * NEG_LOG2E);
  return __builtin_amdgcn_rcpf(1.0f + e);
}

// Opaque zero: redefined by empty asm each call so weight-fragment loads are not
// loop-invariant (defeats LICM/PRE) while keeping LDS addrspace inference.
__device__ __forceinline__ int opaque_zero() {
  int z = 0;
  asm volatile("" : "+s"(z));
  return z;
}

// B fragment (32x16 f16): lane n (0-15) holds column n, K = 0..15; lanes 16-31 K = 16..31.
// Weights stored in LDS g-major, K contiguous (row length KP) -> 32 contiguous bytes per lane.
__device__ __forceinline__ v16h load_bfrag(const _Float16* W, int nt, int kt, int col, int hi) {
  const _Float16* p = W + (nt * 16 + col) * KP + kt * 32 + hi * 16;
  v8h a = *(const v8h*)(p);
  v8h b = *(const v8h*)(p + 8);
  v16h r;
#pragma unroll
  for (int i = 0; i < 8; ++i) { r[i] = a[i]; r[i + 8] = b[i]; }
  return r;
}

// A fragment (16x32 f16) from row-major [16][HP] staging:
// lane m (0-15): halves 0..15 = K 0..15 ; lanes 16-31: K 16..31.
__device__ __forceinline__ v16h load_afrag(const _Float16* st, int col, int hi) {
  const _Float16* p = st + col * HP + hi * 16;
  v8h a = *(const v8h*)(p);
  v8h b = *(const v8h*)(p + 8);
  v16h r;
#pragma unroll
  for (int i = 0; i < 8; ++i) { r[i] = a[i]; r[i + 8] = b[i]; }
  return r;
}

__global__ __launch_bounds__(256)
void fused_lstm2_fc_kernel(const float* __restrict__ x,
                           const float* __restrict__ W_ih0, const float* __restrict__ W_hh0,
                           const float* __restrict__ b_ih0, const float* __restrict__ b_hh0,
                           const float* __restrict__ W_ih1, const float* __restrict__ W_hh1,
                           const float* __restrict__ b_ih1, const float* __restrict__ b_hh1,
                           const float* __restrict__ W_fc1, const float* __restrict__ b_fc1,
                           const float* __restrict__ W_fc2, const float* __restrict__ b_fc2,
                           const float* __restrict__ W_fc3, const float* __restrict__ b_fc3,
                           float* __restrict__ out) {
  __shared__ __align__(16) _Float16 sW0[GP * KP];            // [Wih0(pad K 0..31) | Whh0(K 32..63)]
  __shared__ __align__(16) _Float16 sW1[GP * KP];            // [Wih1 | Whh1]
  __shared__ float sBias0[GP];
  __shared__ float sBias1[GP];
  __shared__ __align__(16) _Float16 sH[8][2][16 * HP];       // [wave][layer][m*HP + j]

  const int tid  = threadIdx.x;
  const int wave = tid >> 5;
  const int lane = tid & 31;
  const int col  = lane & 15;
  const int hi   = lane >> 4;
  const int rowBase = blockIdx.x * ROWS_PER_WG + wave * 16;

  // ---- Stage padded weights into LDS (once) ----
  for (int i = tid; i < GP * KP; i += 256) {
    int g = i >> 6;            // gate index 0..127
    int k = i & 63;            // K index 0..63
    int gate = g >> 5, j = g & 31;
    float w0 = 0.f, w1 = 0.f;
    if (j < 30) {
      int og = gate * 30 + j;
      if (k < LSTM_IN)            w0 = W_ih0[og * LSTM_IN + k];
      else if (k >= 32 && k < 62) w0 = W_hh0[og * 30 + (k - 32)];
      if (k < 30)                 w1 = W_ih1[og * 30 + k];
      else if (k >= 32 && k < 62) w1 = W_hh1[og * 30 + (k - 32)];
    }
    sW0[i] = (_Float16)w0;
    sW1[i] = (_Float16)w1;
  }
  for (int g = tid; g < GP; g += 256) {
    int gate = g >> 5, j = g & 31;
    float bb0 = 0.f, bb1 = 0.f;
    if (j < 30) {
      int og = gate * 30 + j;
      bb0 = b_ih0[og] + b_hh0[og];
      bb1 = b_ih1[og] + b_hh1[og];
    }
    sBias0[g] = bb0;
    sBias1[g] = bb1;
  }
  __syncthreads();

  // Pre-scaled bias per N-tile: i/f/o tiles (nt 0-3,6-7) scaled by -log2e for sigmoid,
  // g tiles (nt 4,5) scaled by -2log2e for tanh. Folds the bias add into the exp2 fma.
  float bias0s[8], bias1s[8];
#pragma unroll
  for (int nt = 0; nt < 8; ++nt) {
    float sc = (nt == 4 || nt == 5) ? NEG_2LOG2E : NEG_LOG2E;
    bias0s[nt] = sBias0[nt * 16 + col] * sc;
    bias1s[nt] = sBias1[nt * 16 + col] * sc;
  }

  _Float16* st0 = &sH[wave][0][0];
  _Float16* st1 = &sH[wave][1][0];

  // Shared zero C tuple for the first WMMA of every accumulation chain (read-only).
  v8f zc;
#pragma unroll
  for (int v = 0; v < 8; ++v) zc[v] = 0.f;

  // ---- Persistent per-wave state ----
  v8f c0[2], c1[2];
#pragma unroll
  for (int ht = 0; ht < 2; ++ht)
#pragma unroll
    for (int v = 0; v < 8; ++v) { c0[ht][v] = 0.f; c1[ht][v] = 0.f; }

  v16h h0frag, h1frag, xfrag;
#pragma unroll
  for (int i = 0; i < 16; ++i) {
    h0frag[i] = (_Float16)0.f; h1frag[i] = (_Float16)0.f; xfrag[i] = (_Float16)0.f;
  }

  // x prefetch (lane's row = rowBase+col; duplicated rows for hi==1 lanes hit zero weight rows)
  const float* xrow = x + (size_t)(rowBase + col) * LSTM_T * LSTM_IN;
  float nx0 = xrow[0], nx1 = xrow[1], nx2 = xrow[2];

  // ---- Barrier-free recurrence: each wave only touches its own staging rows ----
#pragma clang loop unroll(disable)
  for (int t = 0; t < LSTM_T; ++t) {
    const int zz = opaque_zero();            // defeats hoisting of weight fragments
    const _Float16* W0p = sW0 + zz;
    const _Float16* W1p = sW1 + zz;

    xfrag[0] = (_Float16)nx0; xfrag[1] = (_Float16)nx1; xfrag[2] = (_Float16)nx2;
    int tn = (t + 1 < LSTM_T) ? t + 1 : t;
    nx0 = xrow[tn * LSTM_IN + 0];
    nx1 = xrow[tn * LSTM_IN + 1];
    nx2 = xrow[tn * LSTM_IN + 2];

    // ===== Layer 0: gates = [x | h0] * W0^T (+ b folded into activations) =====
#pragma unroll
    for (int ht = 0; ht < 2; ++ht) {
      v8f acc[4];
#pragma unroll
      for (int gate = 0; gate < 4; ++gate) {
        int nt = gate * 2 + ht;
        v16h bf0 = load_bfrag(W0p, nt, 0, col, hi);
        v16h bf1 = load_bfrag(W0p, nt, 1, col, hi);
        v8f a;
        a = __builtin_amdgcn_wmma_f32_16x16x32_f16(false, xfrag, false, bf0, (short)0, zc, false, false);
        a = __builtin_amdgcn_wmma_f32_16x16x32_f16(false, h0frag, false, bf1, (short)0, a, false, false);
        acc[gate] = a;
      }
      float bI = bias0s[0 + ht], bF = bias0s[2 + ht], bG = bias0s[4 + ht], bO = bias0s[6 + ht];
#pragma unroll
      for (int v = 0; v < 8; ++v) {
        float ig = sigmoid_b(acc[0][v], bI);
        float fg = sigmoid_b(acc[1][v], bF);
        float gg = tanh_b  (acc[2][v], bG);
        float og = sigmoid_b(acc[3][v], bO);
        float c  = fmaf(fg, c0[ht][v], ig * gg);
        c0[ht][v] = c;
        float h = og * fast_tanh(c);
        st0[(v + hi * 8) * HP + ht * 16 + col] = (_Float16)h;   // C-layout -> row-major (transpose)
      }
    }
    h0frag = load_afrag(st0, col, hi);   // same-wave DS ordering: no barrier needed

    // ===== Layer 1: gates = [h0 | h1] * W1^T (+ b folded into activations) =====
#pragma unroll
    for (int ht = 0; ht < 2; ++ht) {
      v8f acc[4];
#pragma unroll
      for (int gate = 0; gate < 4; ++gate) {
        int nt = gate * 2 + ht;
        v16h bf0 = load_bfrag(W1p, nt, 0, col, hi);
        v16h bf1 = load_bfrag(W1p, nt, 1, col, hi);
        v8f a;
        a = __builtin_amdgcn_wmma_f32_16x16x32_f16(false, h0frag, false, bf0, (short)0, zc, false, false);
        a = __builtin_amdgcn_wmma_f32_16x16x32_f16(false, h1frag, false, bf1, (short)0, a, false, false);
        acc[gate] = a;
      }
      float bI = bias1s[0 + ht], bF = bias1s[2 + ht], bG = bias1s[4 + ht], bO = bias1s[6 + ht];
#pragma unroll
      for (int v = 0; v < 8; ++v) {
        float ig = sigmoid_b(acc[0][v], bI);
        float fg = sigmoid_b(acc[1][v], bF);
        float gg = tanh_b  (acc[2][v], bG);
        float og = sigmoid_b(acc[3][v], bO);
        float c  = fmaf(fg, c1[ht][v], ig * gg);
        c1[ht][v] = c;
        float h = og * fast_tanh(c);
        st1[(v + hi * 8) * HP + ht * 16 + col] = (_Float16)h;
      }
    }
    h1frag = load_afrag(st1, col, hi);
  }

  __syncthreads();   // make every wave's final h1 staging visible to all threads

  // ---- FC head: one row per thread (threads 0..127), reads final h1 (f16) from staging ----
  if (tid < ROWS_PER_WG) {
    const _Float16* hr = &sH[tid >> 4][1][(tid & 15) * HP];
    float hv[30];
#pragma unroll
    for (int k = 0; k < 30; ++k) hv[k] = (float)hr[k];

    float z1[64];
#pragma unroll 8
    for (int o = 0; o < 64; ++o) {
      float s = b_fc1[o];
#pragma unroll
      for (int k = 0; k < 30; ++k) s = fmaf(hv[k], W_fc1[o * 30 + k], s);
      z1[o] = fmaxf(s, 0.f);
    }
    float z2[32];
#pragma unroll 8
    for (int o = 0; o < 32; ++o) {
      float s = b_fc2[o];
#pragma unroll
      for (int k = 0; k < 64; ++k) s = fmaf(z1[k], W_fc2[o * 64 + k], s);
      z2[o] = fmaxf(s, 0.f);
    }
    float s = b_fc3[0];
#pragma unroll
    for (int k = 0; k < 32; ++k) s = fmaf(z2[k], W_fc3[k], s);
    out[blockIdx.x * ROWS_PER_WG + tid] = fast_sigmoid(s);
  }
}

extern "C" void kernel_launch(void* const* d_in, const int* in_sizes, int n_in,
                              void* d_out, int out_size, void* d_ws, size_t ws_size,
                              hipStream_t stream) {
  const float* x     = (const float*)d_in[0];
  const float* W_ih0 = (const float*)d_in[1];
  const float* W_hh0 = (const float*)d_in[2];
  const float* b_ih0 = (const float*)d_in[3];
  const float* b_hh0 = (const float*)d_in[4];
  const float* W_ih1 = (const float*)d_in[5];
  const float* W_hh1 = (const float*)d_in[6];
  const float* b_ih1 = (const float*)d_in[7];
  const float* b_hh1 = (const float*)d_in[8];
  const float* W_fc1 = (const float*)d_in[9];
  const float* b_fc1 = (const float*)d_in[10];
  const float* W_fc2 = (const float*)d_in[11];
  const float* b_fc2 = (const float*)d_in[12];
  const float* W_fc3 = (const float*)d_in[13];
  const float* b_fc3 = (const float*)d_in[14];

  dim3 grid(LSTM_BATCH / ROWS_PER_WG);   // 64 workgroups
  dim3 block(256);                        // 8 waves x 16 batch rows
  fused_lstm2_fc_kernel<<<grid, block, 0, stream>>>(
      x, W_ih0, W_hh0, b_ih0, b_hh0, W_ih1, W_hh1, b_ih1, b_hh1,
      W_fc1, b_fc1, W_fc2, b_fc2, W_fc3, b_fc3, (float*)d_out);
}
